// Attention_70892730187933
// MI455X (gfx1250) — matrix operand order repaired and checked by reading, not compiled, not run on hardware
//
#include <hip/hip_runtime.h>
#include <hip/hip_bf16.h>
#include <math.h>

// Problem constants
#define NQ    2048
#define DIMC  1024
#define HEADS 16
#define DH    64
#define NN    2
#define KTOT  2050   // keys incl. null kv
#define KTP   2080   // padded to multiple of 32

typedef __bf16 v16bf __attribute__((ext_vector_type(16)));
typedef float  v8f   __attribute__((ext_vector_type(8)));
typedef unsigned int   u32;
typedef unsigned short u16;

union Frag16 {
  v16bf v;
  uint4 q[2];
  u16   h[16];
};

__device__ __forceinline__ u16 f2bf(float f) {
  u32 x = __float_as_uint(f);
  u32 r = x + 0x7fffu + ((x >> 16) & 1u);   // round-to-nearest-even
  return (u16)(r >> 16);
}

// A-matrix 16x32 bf16 fragment (ISA 7.12.2): lane m%16, group g=lane/16.
// Lane holds two contiguous 8-elem chunks: k = kb+8g .. +7 and k = kb+16+8g .. +7
__device__ __forceinline__ v16bf load_a_frag(const u16* base, int ld, int row0, int kb, int lane) {
  const int m = lane & 15, g = lane >> 4;
  const u16* p = base + (size_t)(row0 + m) * ld + kb + 8 * g;
  Frag16 f;
  f.q[0] = *(const uint4*)(p);
  f.q[1] = *(const uint4*)(p + 16);
  return f.v;
}

// B-matrix 32x16 bf16 fragment: lane holds column n = lane%16, K = kb+8g .. +7 and kb+16+8g .. +7
// (baseT is "column-major made contiguous": baseT[n][k], row stride ld)
__device__ __forceinline__ v16bf load_b_frag(const u16* baseT, int ld, int n0, int kb, int lane) {
  const int n = lane & 15, g = lane >> 4;
  const u16* p = baseT + (size_t)(n0 + n) * ld + kb + 8 * g;
  Frag16 f;
  f.q[0] = *(const uint4*)(p);
  f.q[1] = *(const uint4*)(p + 16);
  return f.v;
}

__device__ __forceinline__ v8f wmma_bf16(v16bf a, v16bf b, v8f c) {
  return __builtin_amdgcn_wmma_f32_16x16x32_bf16(false, a, false, b, (short)0, c, false, false);
}

// ---------------------------------------------------------------------------
// LayerNorm + bf16 casts: xn = LN(x)*gamma (bf16), xb = bf16(x)
// ---------------------------------------------------------------------------
__global__ __launch_bounds__(256) void ln_cast_kernel(const float* __restrict__ x,
                                                      const float* __restrict__ gamma,
                                                      u16* __restrict__ xb,
                                                      u16* __restrict__ xn) {
  __shared__ float red[256];
  const int row = blockIdx.x;
  const int tid = threadIdx.x;
  const float* xr = x + (size_t)row * DIMC;

  float v0[4];
  float s = 0.f;
#pragma unroll
  for (int i = 0; i < 4; ++i) { v0[i] = xr[tid + 256 * i]; s += v0[i]; }
  red[tid] = s; __syncthreads();
  for (int w = 128; w > 0; w >>= 1) { if (tid < w) red[tid] += red[tid + w]; __syncthreads(); }
  const float mu = red[0] * (1.0f / (float)DIMC);
  __syncthreads();

  float vs = 0.f;
#pragma unroll
  for (int i = 0; i < 4; ++i) { float d = v0[i] - mu; vs += d * d; }
  red[tid] = vs; __syncthreads();
  for (int w = 128; w > 0; w >>= 1) { if (tid < w) red[tid] += red[tid + w]; __syncthreads(); }
  const float var = red[0] * (1.0f / (float)DIMC);
  const float inv = rsqrtf(var + 1e-5f);

#pragma unroll
  for (int i = 0; i < 4; ++i) {
    const int c = tid + 256 * i;
    xb[(size_t)row * DIMC + c] = f2bf(v0[i]);
    xn[(size_t)row * DIMC + c] = f2bf((v0[i] - mu) * inv * gamma[c]);
  }
}

// ---------------------------------------------------------------------------
// Transpose + cast weights: WT[n*K + k] = bf16(W[k*Nw + n])
// ---------------------------------------------------------------------------
__global__ __launch_bounds__(256) void transpose_cast(const float* __restrict__ W,
                                                      u16* __restrict__ WT,
                                                      int K, int Nw) {
  const size_t idx = (size_t)blockIdx.x * 256 + threadIdx.x;
  if (idx >= (size_t)K * Nw) return;
  const int n = (int)(idx % Nw);
  const int k = (int)(idx / Nw);
  WT[(size_t)n * K + k] = f2bf(W[idx]);
}

// ---------------------------------------------------------------------------
// Null-KV rows, zero padding, padded key mask
// ---------------------------------------------------------------------------
__global__ __launch_bounds__(256) void prep_kv_pad(const float* __restrict__ null_kv,
                                                   const unsigned char* __restrict__ mask,
                                                   u16* __restrict__ kbuf,
                                                   u16* __restrict__ vT,
                                                   int* __restrict__ maskp) {
  const int t = blockIdx.x * 256 + threadIdx.x;
  if (t < NN * DH) {             // null rows 0..1
    const int r = t / DH, d = t % DH;
    kbuf[(size_t)r * DH + d]      = f2bf(null_kv[(size_t)0 * NN * DH + r * DH + d]);
    vT[(size_t)d * KTP + r]       = f2bf(null_kv[(size_t)1 * NN * DH + r * DH + d]);
  }
  if (t < (KTP - KTOT) * DH) {   // zero pad rows 2050..2079
    const int r = KTOT + t / DH, d = t % DH;
    kbuf[(size_t)r * DH + d] = 0;
    vT[(size_t)d * KTP + r]  = 0;
  }
  if (t < KTP) {
    maskp[t] = (t < NN) ? 1 : ((t < KTOT) ? (mask[t - NN] != 0) : 0);
  }
}

// ---------------------------------------------------------------------------
// Generic bf16 GEMM, one 16x16 tile per wave, K in steps of 32.
// mode 0: fp32 store    mode 1: scaled bf16 store    mode 2: K/V split store
// ---------------------------------------------------------------------------
__global__ __launch_bounds__(128) void gemm_bf16(const u16* __restrict__ A, int lda,
                                                 const u16* __restrict__ BT, int ldb,
                                                 int M, int Nc, int K,
                                                 int mode, float scale,
                                                 float* __restrict__ Cf, int ldc,
                                                 u16* __restrict__ Cb,
                                                 u16* __restrict__ vTout) {
  const int lane = threadIdx.x & 31;
  const int wave = threadIdx.x >> 5;
  const int tiles_n = Nc >> 4;
  const int tiles_total = (M >> 4) * tiles_n;
  const int tile = blockIdx.x * 4 + wave;
  if (tile >= tiles_total) return;   // wave-uniform exit: EXEC all-1 for WMMA

  const int tm = (tile / tiles_n) << 4;
  const int tn = (tile % tiles_n) << 4;

  v8f acc = {};
  for (int kb = 0; kb < K; kb += 32) {
    v16bf a = load_a_frag(A, lda, tm, kb, lane);
    v16bf b = load_b_frag(BT, ldb, tn, kb, lane);
    acc = wmma_bf16(a, b, acc);
  }

  const int n = lane & 15, g = lane >> 4;
#pragma unroll
  for (int j = 0; j < 8; ++j) {
    const int row = tm + j + 8 * g;
    const int col = tn + n;
    const float v = acc[j] * scale;
    if (mode == 0) {
      Cf[(size_t)row * ldc + col] = v;
    } else if (mode == 1) {
      Cb[(size_t)row * ldc + col] = f2bf(v);
    } else {                       // KV epilogue: cols 0..63 -> K, 64..127 -> V^T
      if (col < DH) Cb[(size_t)(row + NN) * DH + col] = f2bf(v);
      else          vTout[(size_t)(col - DH) * KTP + (row + NN)] = f2bf(v);
    }
  }
}

// ---------------------------------------------------------------------------
// Flash attention: one wave per (head, 16-query-row tile). 32 keys/iteration.
// ---------------------------------------------------------------------------
__global__ __launch_bounds__(32) void attn_kernel(const u16* __restrict__ q,     // [2048][1024] bf16 (pre-scaled)
                                                  const u16* __restrict__ kbuf,  // [2080][64] bf16
                                                  const u16* __restrict__ vT,    // [64][2080] bf16
                                                  const float* __restrict__ bias,// [16][2048][2048] f32
                                                  const int* __restrict__ maskp, // [2080]
                                                  u16* __restrict__ out) {       // [2048][1024] bf16
  const int lane = threadIdx.x;
  const int h  = blockIdx.x >> 7;       // 128 query tiles per head
  const int qt = blockIdx.x & 127;
  const int n = lane & 15, g = lane >> 4;

  __shared__ __align__(16) u16 p16[16 * 32];

  // Q A-fragments for K-dim 0..31 and 32..63
  const v16bf qa0 = load_a_frag(q, DIMC, qt * 16, h * DH + 0,  lane);
  const v16bf qa1 = load_a_frag(q, DIMC, qt * 16, h * DH + 32, lane);

  v8f o0 = {}, o1 = {}, o2 = {}, o3 = {};
  float rowmax[8], rowsum[8];
#pragma unroll
  for (int j = 0; j < 8; ++j) { rowmax[j] = -3.0e38f; rowsum[j] = 0.f; }

  const int jmax = qt * 16 + 15 + NN;              // last visible padded key index
  const int jend = ((jmax / 32) + 1) * 32;         // key loop bound (multiple of 32)
  const float* biash = bias + (size_t)h * NQ * NQ;

  for (int j0 = 0; j0 < jend; j0 += 32) {
    // S = Q @ K^T for 32 keys -> two 16-col f32 fragments
    v16bf kb0 = load_b_frag(kbuf, DH, j0,      0,  lane);
    v16bf kb1 = load_b_frag(kbuf, DH, j0,      32, lane);
    v16bf kb2 = load_b_frag(kbuf, DH, j0 + 16, 0,  lane);
    v16bf kb3 = load_b_frag(kbuf, DH, j0 + 16, 32, lane);
    v8f s0 = {}, s1 = {};
    s0 = wmma_bf16(qa0, kb0, s0); s0 = wmma_bf16(qa1, kb1, s0);
    s1 = wmma_bf16(qa0, kb2, s1); s1 = wmma_bf16(qa1, kb3, s1);

    // bias (non-temporal stream) + key mask + causal mask
    const int jp0 = j0 + n;
    const int jp1 = j0 + 16 + n;
    const int mv0 = maskp[jp0];
    const int mv1 = maskp[jp1];
    const bool ib0 = (jp0 >= NN) && (jp0 < KTOT);
    const bool ib1 = (jp1 >= NN) && (jp1 < KTOT);
#pragma unroll
    for (int j = 0; j < 8; ++j) {
      const int i = qt * 16 + j + 8 * g;
      float b0 = 0.f, b1 = 0.f;
      if (ib0) b0 = __builtin_nontemporal_load(biash + (size_t)i * NQ + (jp0 - NN));
      if (ib1) b1 = __builtin_nontemporal_load(biash + (size_t)i * NQ + (jp1 - NN));
      const bool vis0 = mv0 && (jp0 <= i + NN);
      const bool vis1 = mv1 && (jp1 <= i + NN);
      s0[j] = vis0 ? (s0[j] + b0) : -3.0e38f;
      s1[j] = vis1 ? (s1[j] + b1) : -3.0e38f;
    }

    // online softmax (16-lane reductions within each half-wave)
    float alpha[8];
#pragma unroll
    for (int j = 0; j < 8; ++j) {
      float mx = fmaxf(s0[j], s1[j]);
      for (int d = 1; d < 16; d <<= 1) mx = fmaxf(mx, __shfl_xor(mx, d, 32));
      const float nm = fmaxf(rowmax[j], mx);
      alpha[j] = __expf(rowmax[j] - nm);
      rowmax[j] = nm;
      const float p0 = __expf(s0[j] - nm);
      const float p1 = __expf(s1[j] - nm);
      s0[j] = p0; s1[j] = p1;
      float ps = p0 + p1;
      for (int d = 1; d < 16; d <<= 1) ps += __shfl_xor(ps, d, 32);
      rowsum[j] = rowsum[j] * alpha[j] + ps;
      o0[j] *= alpha[j]; o1[j] *= alpha[j]; o2[j] *= alpha[j]; o3[j] *= alpha[j];
    }

    // P: C-layout -> LDS tile -> A-layout (16x32 bf16)
#pragma unroll
    for (int j = 0; j < 8; ++j) {
      p16[(j + 8 * g) * 32 + n]      = f2bf(s0[j]);
      p16[(j + 8 * g) * 32 + 16 + n] = f2bf(s1[j]);
    }
    asm volatile("s_wait_dscnt 0" ::: "memory");   // LDS stores visible before reload
    const v16bf pa = load_a_frag(p16, 32, 0, 0, lane);

    // O += P @ V   (V^T rows are contiguous per output-dim column)
    v16bf vb0 = load_b_frag(vT, KTP, 0,  j0, lane);
    v16bf vb1 = load_b_frag(vT, KTP, 16, j0, lane);
    v16bf vb2 = load_b_frag(vT, KTP, 32, j0, lane);
    v16bf vb3 = load_b_frag(vT, KTP, 48, j0, lane);
    o0 = wmma_bf16(pa, vb0, o0);
    o1 = wmma_bf16(pa, vb1, o1);
    o2 = wmma_bf16(pa, vb2, o2);
    o3 = wmma_bf16(pa, vb3, o3);
    asm volatile("s_wait_dscnt 0" ::: "memory");   // pa consumed before next overwrite
  }

  // normalize + store bf16 [i][h*64 + d]
#pragma unroll
  for (int j = 0; j < 8; ++j) {
    const float inv = 1.0f / rowsum[j];            // null keys guarantee rowsum > 0
    const int i = qt * 16 + j + 8 * g;
    const size_t base = (size_t)i * DIMC + h * DH;
    out[base + 0  + n] = f2bf(o0[j] * inv);
    out[base + 16 + n] = f2bf(o1[j] * inv);
    out[base + 32 + n] = f2bf(o2[j] * inv);
    out[base + 48 + n] = f2bf(o3[j] * inv);
  }
}

// ---------------------------------------------------------------------------
extern "C" void kernel_launch(void* const* d_in, const int* in_sizes, int n_in,
                              void* d_out, int out_size, void* d_ws, size_t ws_size,
                              hipStream_t stream) {
  const float* x       = (const float*)d_in[0];
  const float* bias    = (const float*)d_in[1];
  const float* Wq      = (const float*)d_in[2];
  const float* Wkv     = (const float*)d_in[3];
  const float* null_kv = (const float*)d_in[4];
  const float* Wout    = (const float*)d_in[5];
  const float* gamma   = (const float*)d_in[6];
  const unsigned char* mask = (const unsigned char*)d_in[7];  // bool array, 1 byte/elem
  float* out = (float*)d_out;

  char* ws = (char*)d_ws;
  const size_t MB = 1u << 20;
  u16* xb    = (u16*)(ws + 0);                       // [2048][1024] bf16  (4 MB)
  u16* xn    = (u16*)(ws + 4 * MB);                  // [2048][1024] bf16  (4 MB)
  u16* qbuf  = (u16*)(ws + 8 * MB);                  // [2048][1024] bf16  (4 MB)
  u16* aout  = (u16*)(ws + 12 * MB);                 // [2048][1024] bf16  (4 MB)
  u16* WqT   = (u16*)(ws + 16 * MB);                 // [1024][1024] bf16  (2 MB)
  u16* WoutT = (u16*)(ws + 18 * MB);                 // [1024][1024] bf16  (2 MB)
  u16* WkvT  = (u16*)(ws + 20 * MB);                 // [128][1024]  bf16  (256 KB)
  u16* kbuf  = (u16*)(ws + 20 * MB + 512 * 1024);    // [2080][64]   bf16  (260 KB)
  u16* vT    = (u16*)(ws + 21 * MB);                 // [64][2080]   bf16  (260 KB)
  int* maskp = (int*)(ws + 21 * MB + 512 * 1024);    // [2080] int

  ln_cast_kernel<<<NQ, 256, 0, stream>>>(x, gamma, xb, xn);
  transpose_cast<<<(DIMC * DIMC + 255) / 256, 256, 0, stream>>>(Wq,   WqT,   DIMC, DIMC);
  transpose_cast<<<(DIMC * 2 * DH + 255) / 256, 256, 0, stream>>>(Wkv, WkvT, DIMC, 2 * DH);
  transpose_cast<<<(DIMC * DIMC + 255) / 256, 256, 0, stream>>>(Wout, WoutT, DIMC, DIMC);
  prep_kv_pad<<<(KTP + 255) / 256, 256, 0, stream>>>(null_kv, mask, kbuf, vT, maskp);

  // Q = LN(x) @ Wq * DIM_HEAD^-0.5  (scaled into bf16)
  {
    const int tiles = (NQ / 16) * (DIMC / 16);
    gemm_bf16<<<tiles / 4, 128, 0, stream>>>(xn, DIMC, WqT, DIMC, NQ, DIMC, DIMC,
                                             1, 0.125f, nullptr, DIMC, qbuf, nullptr);
  }
  // [K|V] = x @ Wkv  (K row-major +2 null offset, V transposed)
  {
    const int tiles = (NQ / 16) * ((2 * DH) / 16);
    gemm_bf16<<<tiles / 4, 128, 0, stream>>>(xb, DIMC, WkvT, DIMC, NQ, 2 * DH, DIMC,
                                             2, 1.0f, nullptr, 0, kbuf, vT);
  }
  attn_kernel<<<HEADS * (NQ / 16), 32, 0, stream>>>(qbuf, kbuf, vT, bias, maskp, aout);
  // out = attn_out @ Wout  (fp32)
  {
    const int tiles = (NQ / 16) * (DIMC / 16);
    gemm_bf16<<<tiles / 4, 128, 0, stream>>>(aout, DIMC, WoutT, DIMC, NQ, DIMC, DIMC,
                                             0, 1.0f, out, DIMC, nullptr, nullptr);
  }
}
